// CNN_90546500534707
// MI455X (gfx1250) — matrix operand order repaired and checked by reading, not compiled, not run on hardware
//
#include <hip/hip_runtime.h>
#include <hip/hip_bf16.h>

typedef __attribute__((ext_vector_type(16))) __bf16 v16bf;
typedef __attribute__((ext_vector_type(8)))  __bf16 v8bf;
typedef __attribute__((ext_vector_type(8)))  float  v8f;

union BU { v16bf v; v8bf h[2]; };

static constexpr int NH   = 256;
static constexpr int NW   = 256;
static constexpr int NPIX = NH * NW;      // 65536
static constexpr int CIN  = 256;
static constexpr int KM   = 9;
static constexpr int MS   = 32;

// ---------------- workspace layout (bytes) ----------------
static constexpr size_t O_SUMS     = 0;                      // f32 [9][256]
static constexpr size_t O_COUNTS   = 9216;                   // f32 [9]
static constexpr size_t O_MEAN     = 9472;                   // f32 [9][256]
static constexpr size_t O_VALID    = 18688;                  // i32 [9]
static constexpr size_t O_INVCNT   = 18944;                  // f32 [9]
static constexpr size_t O_COV      = 19200;                  // f32 [9][32][32]
static constexpr size_t O_MASKBITS = 56320;                  // u32 [65536]
static constexpr size_t O_W1P      = 318720;                 // bf16 packed conv1 A-frags
static constexpr size_t O_W2P      = O_W1P + 589824;
static constexpr size_t O_W3P      = O_W2P + 147456;
static constexpr size_t O_FSMBF    = 1097728;                // bf16 [N][256]  (pixel-major)
static constexpr size_t O_C1       = O_FSMBF + 33554432;     // bf16 [N][128]
static constexpr size_t O_C2       = O_C1 + 16777216;        // bf16 [N][64]
static constexpr size_t O_C3       = O_C2 + 8388608;         // bf16 [N][32]

// ---------------- small utility kernels ----------------
__global__ void k_zero(float* p, int n) {
  int i = blockIdx.x * 256 + threadIdx.x;
  if (i < n) p[i] = 0.f;
}

// per-pixel mask bitfield + per-mask counts
__global__ void k_maskbits(const int* __restrict__ masks,
                           unsigned* __restrict__ bits_out,
                           float* __restrict__ counts) {
  __shared__ int scnt[KM];
  int tid = threadIdx.x;
  if (tid < KM) scnt[tid] = 0;
  __syncthreads();
  int n = blockIdx.x * blockDim.x + tid;
  unsigned bits = 0;
  for (int k = 0; k < KM; ++k) {
    int mv = masks[k * NPIX + n];
    if (mv != 0) { bits |= (1u << k); atomicAdd(&scnt[k], 1); }
  }
  bits_out[n] = bits;
  __syncthreads();
  if (tid < KM) atomicAdd(&counts[tid], (float)scnt[tid]);
}

// masked per-channel sums: grid (NPIX/1024, CIN)
__global__ void k_masked_sums(const float* __restrict__ x,
                              const unsigned* __restrict__ bits,
                              float* __restrict__ sums) {
  __shared__ float ssum[KM];
  int tid = threadIdx.x;
  int c = blockIdx.y;
  int base = blockIdx.x * 1024;
  if (tid < KM) ssum[tid] = 0.f;
  __syncthreads();
  float acc[KM];
#pragma unroll
  for (int k = 0; k < KM; ++k) acc[k] = 0.f;
  for (int i = 0; i < 4; ++i) {
    int n = base + i * 256 + tid;
    float v = x[c * NPIX + n];
    unsigned b = bits[n];
#pragma unroll
    for (int k = 0; k < KM; ++k)
      if ((b >> k) & 1u) acc[k] += v;
  }
#pragma unroll
  for (int k = 0; k < KM; ++k) atomicAdd(&ssum[k], acc[k]);
  __syncthreads();
  if (tid < KM) atomicAdd(&sums[tid * CIN + c], ssum[tid]);
}

__global__ void k_means(const float* __restrict__ sums,
                        const float* __restrict__ counts,
                        float* __restrict__ mean, int* __restrict__ valid,
                        float* __restrict__ invcnt) {
  int k = blockIdx.x, c = threadIdx.x;
  float cnt = counts[k];
  mean[k * CIN + c] = sums[k * CIN + c] / fmaxf(cnt, 1.f);
  if (c == 0) {
    valid[k]  = (cnt >= 10.f) ? 1 : 0;
    invcnt[k] = 1.f / fmaxf(cnt, 1.f);
  }
}

// fsm = x - mean[last valid mask]; f32 out stays [C][N], bf16 copy goes [N][C]
__global__ void k_subtract(const float* __restrict__ x,
                           const unsigned* __restrict__ bits,
                           const float* __restrict__ mean,
                           const int* __restrict__ valid,
                           float* __restrict__ fsm, __bf16* __restrict__ fsm_bf) {
  int n = blockIdx.x * blockDim.x + threadIdx.x;
  unsigned b = bits[n];
  int lastk = -1;
  for (int k = KM - 1; k >= 0; --k)
    if (((b >> k) & 1u) && valid[k]) { lastk = k; break; }
  const float* mrow = mean + (lastk >= 0 ? lastk : 0) * CIN;
  for (int c0 = 0; c0 < CIN; c0 += 8) {
    v8bf ob;
#pragma unroll
    for (int j = 0; j < 8; ++j) {
      int c = c0 + j;
      float v = x[c * NPIX + n];
      if (lastk >= 0) v -= mrow[c];
      fsm[c * NPIX + n] = v;
      ob[j] = (__bf16)v;
    }
    *reinterpret_cast<v8bf*>(fsm_bf + (size_t)n * CIN + c0) = ob;
  }
}

// pack f32 OIHW weights into bf16 WMMA A-fragment order:
// [tap][icb][oct][lane][h], K(h,lane) per ISA 16-bit A 16x32 layout
__global__ void k_pack_w(const float* __restrict__ w, __bf16* __restrict__ wp,
                         int IC, int OC) {
  int OCT = OC / 16, ICB = IC / 32;
  int total = 9 * ICB * OCT * 512;
  int e = blockIdx.x * 256 + threadIdx.x;
  if (e >= total) return;
  int h    = e & 15;
  int lane = (e >> 4) & 31;
  int t    = e >> 9;
  int oct = t % OCT; t /= OCT;
  int icb = t % ICB;
  int tap = t / ICB;
  int m = lane & 15;
  int K = ((h & 8) << 1) + ((lane >> 4) << 3) + (h & 7);
  int oc = oct * 16 + m;
  int ic = icb * 32 + K;
  int dy = tap / 3, dx = tap % 3;
  wp[e] = (__bf16)w[((oc * IC + ic) * 3 + dy) * 3 + dx];
}

// ---------------- implicit-GEMM 3x3 SAME conv via WMMA ----------------
// activations pixel-major [N][C]; LDS tile [dy][col][ic(pad 40)];
// double-buffered async DMA: stage slab icb+1 while computing slab icb
template <int IC, int OC, bool RELU>
__global__ __launch_bounds__(256) void k_conv3x3(const __bf16* __restrict__ in,
                                                 const __bf16* __restrict__ wp,
                                                 const float* __restrict__ bias,
                                                 __bf16* __restrict__ out) {
  constexpr int ICB = IC / 32, OCT_ALL = OC / 16;
  constexpr int CP = 40;                     // padded ch stride: 80B (16B aligned, bank-safe)
  __shared__ __align__(16) __bf16 tile[2][3 * 68 * CP];
  int tid  = threadIdx.x;
  int lane = tid & 31;
  int wave = tid >> 5;
  int wc = wave & 3;                         // pixel quadrant (16 px each)
  int wr = wave >> 2;                        // oc half (16 oc each)
  int wtile = blockIdx.x & 3;
  int h     = blockIdx.x >> 2;
  int wbase = wtile * 64;
  int oct   = blockIdx.y * 2 + wr;           // 16-oc tile index

  // stage one 32-channel slab: 3 rows x 68 cols x 32 ch in 16B chunks
  auto stage = [&](int icb, __bf16* buf) {
    for (int e = tid; e < 816; e += 256) {
      int ic8 = (e & 3) * 8;
      int pos = e >> 2;                      // 0..203
      int col = pos % 68;
      int dy  = pos / 68;
      int hpos = h - 1 + dy;
      int wpos = wbase - 1 + col;
      __bf16* lp = &buf[(dy * 68 + col) * CP + ic8];
      if (hpos >= 0 && hpos < NH && wpos >= 0 && wpos < NW) {
        const __bf16* gp = in + ((size_t)(hpos * NW + wpos) * IC + icb * 32 + ic8);
        unsigned ldsoff = (unsigned)(size_t)lp;   // generic addr low 32b = LDS offset
        asm volatile("global_load_async_to_lds_b128 %0, %1, off"
                     :: "v"(ldsoff), "v"(gp) : "memory");
      } else {
        v8bf z = {};
        *reinterpret_cast<v8bf*>(lp) = z;
      }
    }
  };

  v8f c = {};
  stage(0, tile[0]);
  asm volatile("s_wait_asynccnt 0x0" ::: "memory");
  __syncthreads();
  int cur = 0;
  for (int icb = 0; icb < ICB; ++icb) {
    if (icb + 1 < ICB) stage(icb + 1, tile[cur ^ 1]);   // DMA overlaps WMMA below
    const __bf16* tb = tile[cur];
#pragma unroll
    for (int tap = 0; tap < 9; ++tap) {
      int dy = tap / 3, dx = tap % 3;
      int col = wc * 16 + (lane & 15) + dx;
      int coff = (dy * 68 + col) * CP + ((lane & 16) >> 1);    // B: K table order (8h+s, 16+8h+s)
      BU b;
      b.h[0] = *reinterpret_cast<const v8bf*>(&tb[coff]);
      b.h[1] = *reinterpret_cast<const v8bf*>(&tb[coff + 16]);
      const v16bf a = *reinterpret_cast<const v16bf*>(
          wp + ((((size_t)tap * ICB + icb) * OCT_ALL + oct) * 32 + lane) * 16);
      c = __builtin_amdgcn_wmma_f32_16x16x32_bf16(false, a, false, b.v,
                                                  (short)0, c, false, false);
    }
    asm volatile("s_wait_asynccnt 0x0" ::: "memory");   // own async writes done
    __syncthreads();                                    // all waves' writes visible; WAR safe
    cur ^= 1;
  }
  int n   = h * NW + wbase + wc * 16 + (lane & 15);
  int oc0 = oct * 16 + ((lane & 16) >> 1);   // 8 consecutive oc per lane
  v8bf ov;
#pragma unroll
  for (int r = 0; r < 8; ++r) {
    float v = c[r] + bias[oc0 + r];
    if (RELU) v = fmaxf(v, 0.f);
    ov[r] = (__bf16)v;
  }
  *reinterpret_cast<v8bf*>(out + (size_t)n * OC + oc0) = ov;
}

// ---------------- masked covariance C = F_masked * F_masked^T (WMMA) ----------------
// f is [N][32] pixel-major; fm LDS [px][ch(pad 40)]: staging is fully vectorized
// (1x global_load_b128 + 1x ds_store_b128 per thread per step), transpose cost
// moved into the per-wave fragment gathers.
__global__ __launch_bounds__(128) void k_cov(const __bf16* __restrict__ f,
                                             const unsigned* __restrict__ bits,
                                             float* __restrict__ cov) {
  __shared__ __align__(16) __bf16 fm[32][40];          // [px][ch]
  int tid  = threadIdx.x;
  int lane = tid & 31;
  int wave = tid >> 5;
  int c0 = (wave & 1) * 16;
  int d0 = (wave >> 1) * 16;
  int k  = blockIdx.y;
  int chunk0 = blockIdx.x * 4096;
  int spx = tid >> 2;                                  // staging pixel 0..31
  int sch = (tid & 3) * 8;                             // staging channel group
  v8f acc = {};
  for (int n0 = chunk0; n0 < chunk0 + 4096; n0 += 32) {
    __syncthreads();
    {
      int n = n0 + spx;
      v8bf fv = *reinterpret_cast<const v8bf*>(f + (size_t)n * MS + sch);
      unsigned on = (bits[n] >> k) & 1u;
      v8bf ov;
#pragma unroll
      for (int j = 0; j < 8; ++j) ov[j] = on ? fv[j] : (__bf16)0.f;
      *reinterpret_cast<v8bf*>(&fm[spx][sch]) = ov;
    }
    __syncthreads();
    v16bf a, b;
#pragma unroll
    for (int hh = 0; hh < 16; ++hh) {
      int Ka = ((hh & 8) << 1) + ((lane >> 4) << 3) + (hh & 7);
      a[hh] = fm[Ka][c0 + (lane & 15)];                // A(m,K)=F[c0+m][pxK]
      b[hh] = fm[Ka][d0 + (lane & 15)];  // B(K,n)=F[d0+n][pxK]
    }
    acc = __builtin_amdgcn_wmma_f32_16x16x32_bf16(false, a, false, b,
                                                  (short)0, acc, false, false);
  }
  int col  = d0 + (lane & 15);
  int row0 = c0 + ((lane & 16) >> 1);
#pragma unroll
  for (int r = 0; r < 8; ++r)
    atomicAdd(&cov[(k * MS + row0 + r) * MS + col], acc[r]);
}

// trans[k] = (cov[k]/cnt) @ fc_w^T + fc_b, zeroed when cnt < 10
__global__ void k_fc(const float* __restrict__ cov, const float* __restrict__ fc_w,
                     const float* __restrict__ fc_b, const int* __restrict__ valid,
                     const float* __restrict__ invcnt, float* __restrict__ trans) {
  int k = blockIdx.x;
  int o = blockIdx.y * 256 + threadIdx.x;
  float s = 0.f;
  if (valid[k]) {
    float inv = invcnt[k];
    s = fc_b[o];
    const float* cr = cov + k * 1024;
    const float* wr = fc_w + (size_t)o * 1024;
    for (int j = 0; j < 1024; ++j) s += cr[j] * inv * wr[j];
  }
  trans[k * 1024 + o] = s;
}

// ---------------- launch ----------------
extern "C" void kernel_launch(void* const* d_in, const int* in_sizes, int n_in,
                              void* d_out, int out_size, void* d_ws, size_t ws_size,
                              hipStream_t stream) {
  const float* x     = (const float*)d_in[0];
  const int*   masks = (const int*)d_in[1];
  const float* w1    = (const float*)d_in[2];
  const float* b1    = (const float*)d_in[3];
  const float* w2    = (const float*)d_in[4];
  const float* b2    = (const float*)d_in[5];
  const float* w3    = (const float*)d_in[6];
  const float* b3    = (const float*)d_in[7];
  const float* fc_w  = (const float*)d_in[8];
  const float* fc_b  = (const float*)d_in[9];

  float* trans_out = (float*)d_out;
  float* fsm_out   = trans_out + KM * 1024;

  char* ws = (char*)d_ws;
  float*    sums   = (float*)(ws + O_SUMS);
  float*    counts = (float*)(ws + O_COUNTS);
  float*    mean   = (float*)(ws + O_MEAN);
  int*      valid  = (int*)(ws + O_VALID);
  float*    invcnt = (float*)(ws + O_INVCNT);
  float*    cov    = (float*)(ws + O_COV);
  unsigned* mbits  = (unsigned*)(ws + O_MASKBITS);
  __bf16*   w1p    = (__bf16*)(ws + O_W1P);
  __bf16*   w2p    = (__bf16*)(ws + O_W2P);
  __bf16*   w3p    = (__bf16*)(ws + O_W3P);
  __bf16*   fsm_bf = (__bf16*)(ws + O_FSMBF);
  __bf16*   c1     = (__bf16*)(ws + O_C1);
  __bf16*   c2     = (__bf16*)(ws + O_C2);
  __bf16*   c3     = (__bf16*)(ws + O_C3);

  k_zero<<<10, 256, 0, stream>>>(sums, 2304 + 9);
  k_zero<<<36, 256, 0, stream>>>(cov, KM * MS * MS);

  k_maskbits<<<NPIX / 256, 256, 0, stream>>>(masks, mbits, counts);
  k_masked_sums<<<dim3(NPIX / 1024, CIN), 256, 0, stream>>>(x, mbits, sums);
  k_means<<<KM, CIN, 0, stream>>>(sums, counts, mean, valid, invcnt);
  k_subtract<<<NPIX / 256, 256, 0, stream>>>(x, mbits, mean, valid, fsm_out, fsm_bf);

  k_pack_w<<<(9 * 8 * 8 * 512 + 255) / 256, 256, 0, stream>>>(w1, w1p, 256, 128);
  k_pack_w<<<(9 * 4 * 4 * 512 + 255) / 256, 256, 0, stream>>>(w2, w2p, 128, 64);
  k_pack_w<<<(9 * 2 * 2 * 512 + 255) / 256, 256, 0, stream>>>(w3, w3p, 64, 32);

  k_conv3x3<256, 128, true ><<<dim3(1024, 4), 256, 0, stream>>>(fsm_bf, w1p, b1, c1);
  k_conv3x3<128, 64,  true ><<<dim3(1024, 2), 256, 0, stream>>>(c1, w2p, b2, c2);
  k_conv3x3<64,  32,  false><<<dim3(1024, 1), 256, 0, stream>>>(c2, w3p, b3, c3);

  k_cov<<<dim3(16, KM), 128, 0, stream>>>(c3, mbits, cov);
  k_fc<<<dim3(KM, 4), 256, 0, stream>>>(cov, fc_w, fc_b, valid, invcnt, trans_out);
}
